// MGU_1108101562913
// MI455X (gfx1250) — matrix operand order repaired and checked by reading, not compiled, not run on hardware
//
#include <hip/hip_runtime.h>

// MGU: h_t = (1-f)h + f*tanh(xh_t + (f*h)@Uh + bh), f = sigmoid(xf_t + h@Uf + bf)
// B=128, T=512, D=U=512.
//  1) weights -> transposed (N-major) bf16, x -> bf16
//  2) fused dual GEMM (WMMA bf16): xf, xh in time-major [T,B,U] f32.
//     Per block: x tile staged in LDS via Tensor Data Mover; fully unrolled
//     K loop with explicit distance-1 prefetch (SSA ping-pong, no reg copies).
//  3) per-timestep gate/candidate WMMA kernels: weight tile staged in LDS via
//     TDM (shared by 4 waves); fully unrolled K loop with distance-2 prefetch.

#define B_DIM 128
#define T_DIM 512
#define D_DIM 512
#define U_DIM 512

typedef __bf16 bf16;
typedef __attribute__((ext_vector_type(16))) __bf16 v16bf;
typedef __attribute__((ext_vector_type(8)))  float  v8f;

typedef __attribute__((ext_vector_type(4))) unsigned int tdm_g0_t;
typedef __attribute__((ext_vector_type(8))) int          tdm_g1_t;
typedef __attribute__((ext_vector_type(4))) int          tdm_g2_t;
typedef __attribute__((ext_vector_type(8))) int          tdm_g4_t;

#if defined(__has_builtin)
#if __has_builtin(__builtin_amdgcn_tensor_load_to_lds) && \
    __has_builtin(__builtin_amdgcn_s_wait_tensorcnt)
#define HAVE_TDM 1
#endif
#endif

__device__ __forceinline__ bf16 f2bf(float f) {
  union { float f; unsigned u; } in;
  in.f = f;
  unsigned u = in.u;
  unsigned r = (u + 0x7FFFu + ((u >> 16) & 1u)) >> 16;  // round-to-nearest-even
  union { unsigned short s; bf16 b; } out;
  out.s = (unsigned short)r;
  return out.b;
}

#ifdef HAVE_TDM
// 1D contiguous TDM copy: `elems` bf16 elements global -> LDS.
// D# per CDNA5 ISA 8.3/8.4: count=1, type=2(image), data_size=1(2B),
// tile_dim0=elems, tile_dim1=1, tensor dims sized to the tile (no OOB).
__device__ __forceinline__ void tdm_load_1d(unsigned lds_addr, const void* gptr,
                                            unsigned elems) {
  unsigned long long ga = (unsigned long long)(size_t)gptr;
  tdm_g0_t g0;
  g0[0] = 1u;                                      // count=1, is_restore=0
  g0[1] = lds_addr;                                // lds_addr (bytes)
  g0[2] = (unsigned)(ga & 0xFFFFFFFFu);            // global_addr[31:0]
  g0[3] = (unsigned)((ga >> 32) & 0x01FFFFFFu)     // global_addr[56:32]
          | 0x80000000u;                           // type=2 at bits[127:126]
  tdm_g1_t g1;
  g1[0] = 0x00010000;                              // data_size=1 (2B), mask=0
  g1[1] = (int)((elems & 0xFFFFu) << 16);          // tensor_dim0[15:0]
  g1[2] = (int)(((elems >> 16) & 0xFFFFu)          // tensor_dim0[31:16]
                | (1u << 16));                     // tensor_dim1 = 1
  g1[3] = (int)((elems & 0xFFFFu) << 16);          // tile_dim0 = elems
  g1[4] = 1;                                       // tile_dim1 = 1
  g1[5] = (int)elems;                              // tensor_dim0_stride[31:0]
  g1[6] = 0;
  g1[7] = 0;
  tdm_g2_t gz = {0, 0, 0, 0};
  tdm_g4_t gz8 = {0, 0, 0, 0, 0, 0, 0, 0};
  __builtin_amdgcn_tensor_load_to_lds(g0, g1, gz, gz, gz8, 0);
}
#endif

// Stage `elems` bf16 global -> LDS for the whole block, then barrier.
__device__ __forceinline__ void stage_to_lds(bf16* lds, const bf16* g, int elems) {
#ifdef HAVE_TDM
  if ((threadIdx.x >> 5) == 0) {
    tdm_load_1d((unsigned)(size_t)(void*)lds, (const void*)g, (unsigned)elems);
    __builtin_amdgcn_s_wait_tensorcnt(0);
  }
#else
  const int nvec = elems >> 3;  // uint4 = 8 bf16
  for (int i = threadIdx.x; i < nvec; i += blockDim.x)
    reinterpret_cast<uint4*>(lds)[i] = reinterpret_cast<const uint4*>(g)[i];
#endif
  __syncthreads();
}

__device__ __forceinline__ v16bf load_frag_2x8(const bf16* p0, const bf16* p1) {
  union { v16bf v; uint4 q[2]; } u;
  u.q[0] = *reinterpret_cast<const uint4*>(p0);
  u.q[1] = *reinterpret_cast<const uint4*>(p1);
  return u.v;
}

// A fragment: 16x32 (MxK), row-major bf16, leading dim ld.
// lanes 0-15 row M=lane hold K {0..7,16..23}; lanes 16-31 hold K {8..15,24..31}.
__device__ __forceinline__ v16bf load_a(const bf16* A, int ld, int m0, int k0, int lane) {
  int r = lane & 15, h = lane >> 4;
  const bf16* base = A + (m0 + r) * ld + k0 + h * 8;
  return load_frag_2x8(base, base + 16);
}

// B fragment: 32x16 (KxN) from N-major weights Bt[n][k], leading dim ld.
// lanes 0-15 col N=lane hold K {0..7,16..23}; lanes 16-31 hold K {8..15,24..31}.
__device__ __forceinline__ v16bf load_bt(const bf16* Bt, int ld, int n0, int k0, int lane) {
  int r = lane & 15, h = lane >> 4;
  const bf16* base = Bt + (n0 + r) * ld + k0 + h * 8;
  return load_frag_2x8(base, base + 16);
}

__device__ __forceinline__ v8f wmma_bf16(v16bf a, v16bf b, v8f c) {
  return __builtin_amdgcn_wmma_f32_16x16x32_bf16(false, a, false, b, (short)0, c,
                                                 false, false);
}

// ---- conversion kernels -----------------------------------------------------

__global__ void transpose_cvt_kernel(const float* __restrict__ in,
                                     bf16* __restrict__ out, int rows, int cols) {
  int idx = blockIdx.x * blockDim.x + threadIdx.x;
  if (idx >= rows * cols) return;
  int n = idx / rows;
  int k = idx - n * rows;
  out[n * rows + k] = f2bf(in[k * cols + n]);
}

__global__ void cvt_bf16_kernel(const float* __restrict__ in,
                                bf16* __restrict__ out, int n) {
  int idx = blockIdx.x * blockDim.x + threadIdx.x;
  if (idx < n) out[idx] = f2bf(in[idx]);
}

// ---- fused input projections: XF/XH[T,B,U] = x[B*T,D] @ {Wf,Wh} ------------
// Block (128 thr) = one 16-row M tile (x tile in LDS via TDM) x 128 columns.
// Wave w handles 2 N-tiles for BOTH matrices (4 accumulators). Fully unrolled
// K loop with explicit distance-1 prefetch of the 4 B fragments per stage.
__global__ __launch_bounds__(128) void precompute_kernel(
    const bf16* __restrict__ Xb,   // [B*T, D] bf16
    const bf16* __restrict__ Wft,  // [U, D]   bf16 (N-major)
    const bf16* __restrict__ Wht,  // [U, D]   bf16 (N-major)
    float* __restrict__ XF,        // [T, B, U] f32
    float* __restrict__ XH) {      // [T, B, U] f32
  __shared__ __align__(16) bf16 As[16 * D_DIM];   // 16KB x tile

  int mtile = blockIdx.x >> 2;     // 0..4095
  int q     = blockIdx.x & 3;      // column quarter (128 cols)
  int warp  = threadIdx.x >> 5;
  int lane  = threadIdx.x & 31;
  int m0    = mtile * 16;
  int n0    = q * 128 + warp * 32; // this wave: columns n0..n0+31

  stage_to_lds(As, Xb + (size_t)m0 * D_DIM, 16 * D_DIM);

  v8f accF[2] = {};
  v8f accH[2] = {};
  v16bf bFc[2], bHc[2];
#pragma unroll
  for (int j = 0; j < 2; ++j) {
    bFc[j] = load_bt(Wft, D_DIM, n0 + j * 16, 0, lane);
    bHc[j] = load_bt(Wht, D_DIM, n0 + j * 16, 0, lane);
  }
#pragma unroll
  for (int k = 0; k < D_DIM; k += 32) {
    int kn = k + 32;
    v16bf bFn[2], bHn[2];
    if (kn < D_DIM) {               // static under full unroll
#pragma unroll
      for (int j = 0; j < 2; ++j) {
        bFn[j] = load_bt(Wft, D_DIM, n0 + j * 16, kn, lane);
        bHn[j] = load_bt(Wht, D_DIM, n0 + j * 16, kn, lane);
      }
    }
    v16bf a = load_a(As, D_DIM, 0, k, lane);   // LDS
#pragma unroll
    for (int j = 0; j < 2; ++j) {
      accF[j] = wmma_bf16(a, bFc[j], accF[j]);
      accH[j] = wmma_bf16(a, bHc[j], accH[j]);
    }
#pragma unroll
    for (int j = 0; j < 2; ++j) { bFc[j] = bFn[j]; bHc[j] = bHn[j]; }  // SSA rename
  }

  int r = lane & 15, h = lane >> 4;
#pragma unroll
  for (int j = 0; j < 2; ++j) {
    int u = n0 + j * 16 + r;
#pragma unroll
    for (int v = 0; v < 8; ++v) {
      int m = m0 + h * 8 + v;        // row of x[B*T]: m = b*T + t
      int b = m >> 9;
      int t = m & (T_DIM - 1);
      int o = (t * B_DIM + b) * U_DIM + u;
      XF[o] = accF[j][v];
      XH[o] = accH[j][v];
    }
  }
}

// ---- recurrent step GEMM core: block = 1 N-tile (weights in LDS) x 4 M-tiles
// Fully unrolled K loop with distance-2 prefetch queue (SSA renames only).
__device__ __forceinline__ v8f step_gemm(const bf16* __restrict__ A,  // [B,U] bf16
                                         const bf16* __restrict__ Bt, // [U,U] N-major
                                         bf16* Bs, int m0, int ntile, int lane) {
  stage_to_lds(Bs, Bt + (size_t)ntile * 16 * U_DIM, 16 * U_DIM);
  v8f acc = {};
  v16bf aQ[2], bQ[2];
  aQ[0] = load_a(A, U_DIM, m0, 0, lane);
  bQ[0] = load_bt(Bs, U_DIM, 0, 0, lane);
  aQ[1] = load_a(A, U_DIM, m0, 32, lane);
  bQ[1] = load_bt(Bs, U_DIM, 0, 32, lane);
#pragma unroll
  for (int k = 0; k < U_DIM; k += 32) {
    v16bf a_use = aQ[0], b_use = bQ[0];
    aQ[0] = aQ[1];                  // SSA rename under full unroll
    bQ[0] = bQ[1];
    int kp = k + 64;
    if (kp < U_DIM) {               // static under full unroll
      aQ[1] = load_a(A, U_DIM, m0, kp, lane);
      bQ[1] = load_bt(Bs, U_DIM, 0, kp, lane);
    }
    acc = wmma_bf16(a_use, b_use, acc);
  }
  return acc;
}

// phase 1: f = sigmoid(xf_t + h@Uf + bf); g = f*h
__global__ __launch_bounds__(128) void step_gate_kernel(
    const bf16* __restrict__ Hb,   // [B,U] bf16 (previous h)
    const float* __restrict__ Hf,  // [B,U] f32  (previous h)
    const bf16* __restrict__ Uft,  // [U,U] bf16 (N-major)
    const float* __restrict__ XF,  // [T,B,U] f32
    const float* __restrict__ bfv, // [U]
    float* __restrict__ F,         // [B,U] f32 gate out
    bf16*  __restrict__ G,         // [B,U] bf16 (f*h) out
    int t) {
  __shared__ __align__(16) bf16 Bs[16 * U_DIM];   // 16KB Uf tile
  int ntile = blockIdx.x >> 1;                    // 0..31
  int mh    = blockIdx.x & 1;                     // M half
  int warp  = threadIdx.x >> 5;
  int lane  = threadIdx.x & 31;
  int mtile = mh * 4 + warp;                      // 0..7
  int m0    = mtile * 16;

  v8f acc = step_gemm(Hb, Uft, Bs, m0, ntile, lane);

  int r = lane & 15, h = lane >> 4;
  int u = ntile * 16 + r;
  float bias = bfv[u];
  const float* xf_t = XF + (size_t)t * (B_DIM * U_DIM);
#pragma unroll
  for (int v = 0; v < 8; ++v) {
    int b_ = m0 + h * 8 + v;
    int o = b_ * U_DIM + u;
    float z = acc[v] + xf_t[o] + bias;
    float f = 1.0f / (1.0f + expf(-z));
    F[o] = f;
    G[o] = f2bf(f * Hf[o]);
  }
}

// phase 2: S = tanh(xh_t + g@Uh + bh); h = (1-f)h + f*S
__global__ __launch_bounds__(128) void step_cand_kernel(
    const bf16* __restrict__ G,    // [B,U] bf16 (f*h)
    const bf16* __restrict__ Uht,  // [U,U] bf16 (N-major)
    const float* __restrict__ XH,  // [T,B,U] f32
    const float* __restrict__ bhv, // [U]
    const float* __restrict__ F,   // [B,U] f32 gate
    float* __restrict__ Hf,        // [B,U] f32 h (in/out, lives in d_out)
    bf16*  __restrict__ Hb,        // [B,U] bf16 h (out)
    int t) {
  __shared__ __align__(16) bf16 Bs[16 * U_DIM];   // 16KB Uh tile
  int ntile = blockIdx.x >> 1;
  int mh    = blockIdx.x & 1;
  int warp  = threadIdx.x >> 5;
  int lane  = threadIdx.x & 31;
  int mtile = mh * 4 + warp;
  int m0    = mtile * 16;

  v8f acc = step_gemm(G, Uht, Bs, m0, ntile, lane);

  int r = lane & 15, h = lane >> 4;
  int u = ntile * 16 + r;
  float bias = bhv[u];
  const float* xh_t = XH + (size_t)t * (B_DIM * U_DIM);
#pragma unroll
  for (int v = 0; v < 8; ++v) {
    int b_ = m0 + h * 8 + v;
    int o = b_ * U_DIM + u;
    float s = tanhf(acc[v] + xh_t[o] + bias);
    float f = F[o];
    float hn = (1.0f - f) * Hf[o] + f * s;
    Hf[o] = hn;
    Hb[o] = f2bf(hn);
  }
}

// ---- driver -----------------------------------------------------------------

extern "C" void kernel_launch(void* const* d_in, const int* in_sizes, int n_in,
                              void* d_out, int out_size, void* d_ws, size_t ws_size,
                              hipStream_t stream) {
  const float* x   = (const float*)d_in[0];  // [B,T,D]
  const float* Wf  = (const float*)d_in[1];  // [D,U]
  const float* Uf  = (const float*)d_in[2];  // [U,U]
  const float* bf_ = (const float*)d_in[3];  // [U]
  const float* Wh  = (const float*)d_in[4];  // [D,U]
  const float* Uh  = (const float*)d_in[5];  // [U,U]
  const float* bh  = (const float*)d_in[6];  // [U]
  float* h_out = (float*)d_out;              // [B,U] f32 — doubles as h_f32 state

  size_t off = 0;
  auto carve = [&](size_t bytes) -> void* {
    void* p = (char*)d_ws + off;
    off += (bytes + 255) & ~(size_t)255;
    return p;
  };
  bf16* Wft = (bf16*)carve((size_t)U_DIM * D_DIM * 2);
  bf16* Wht = (bf16*)carve((size_t)U_DIM * D_DIM * 2);
  bf16* Uft = (bf16*)carve((size_t)U_DIM * U_DIM * 2);
  bf16* Uht = (bf16*)carve((size_t)U_DIM * U_DIM * 2);
  bf16* Hb  = (bf16*)carve((size_t)B_DIM * U_DIM * 2);
  bf16* G   = (bf16*)carve((size_t)B_DIM * U_DIM * 2);
  float* F  = (float*)carve((size_t)B_DIM * U_DIM * 4);
  bf16* Xb  = (bf16*)carve((size_t)B_DIM * T_DIM * D_DIM * 2);
  float* XF = (float*)carve((size_t)T_DIM * B_DIM * U_DIM * 4);
  float* XH = (float*)carve((size_t)T_DIM * B_DIM * U_DIM * 4);

  // 1) weight transpose+convert (f32 -> bf16, N-major), x convert
  {
    int n = D_DIM * U_DIM;
    int blks = (n + 255) / 256;
    transpose_cvt_kernel<<<blks, 256, 0, stream>>>(Wf, Wft, D_DIM, U_DIM);
    transpose_cvt_kernel<<<blks, 256, 0, stream>>>(Wh, Wht, D_DIM, U_DIM);
    transpose_cvt_kernel<<<blks, 256, 0, stream>>>(Uf, Uft, U_DIM, U_DIM);
    transpose_cvt_kernel<<<blks, 256, 0, stream>>>(Uh, Uht, U_DIM, U_DIM);
  }
  {
    int n = B_DIM * T_DIM * D_DIM;
    cvt_bf16_kernel<<<(n + 255) / 256, 256, 0, stream>>>(x, Xb, n);
  }

  // 2) fused input projections: 4096 M-tiles x 4 column-quarters
  precompute_kernel<<<16384, 128, 0, stream>>>(Xb, Wft, Wht, XF, XH);

  // 3) init h = 0 (f32 in d_out, bf16 mirror in ws)
  (void)hipMemsetAsync(h_out, 0, (size_t)B_DIM * U_DIM * 4, stream);
  (void)hipMemsetAsync(Hb, 0, (size_t)B_DIM * U_DIM * 2, stream);

  // 4) sequential scan: 64 blocks per phase (32 N-tiles x 2 M-halves)
  for (int t = 0; t < T_DIM; ++t) {
    step_gate_kernel<<<64, 128, 0, stream>>>(Hb, h_out, Uft, XF, bf_, F, G, t);
    step_cand_kernel<<<64, 128, 0, stream>>>(G, Uht, XH, bh, F, h_out, Hb, t);
  }
}
